// LoopVisibleLSTM_80324478370225
// MI455X (gfx1250) — matrix operand order repaired and checked by reading, not compiled, not run on hardware
//
#include <hip/hip_runtime.h>
#include <hip/hip_bf16.h>

// Bidirectional LSTM  T=2048, B=32, I=H=512  ->  out [T, B, 2H] fp32
//
// Phase 1: pack x, Wih^T, Whh^T into bf16 WMMA fragment layouts.
// Phase 2: x_proj[dir] = x @ Wih^T + b   via v_wmma_f32_16x16x32_bf16,
//          stored fp32 in D-fragment order (direct C-init in phase 3).
// Phase 3: persistent recurrence: 8 WGs/direction; WG wg owns h-tiles
//          [4wg,4wg+4) x both batch-tiles. Its 256KB Whh slice is loaded
//          into LDS ONCE (async-to-LDS), h (32KB) is async-staged into LDS
//          each step; all WMMA operands then come from ds_load_b128.
//          c-state lives in VGPRs; per-step agent-scope barrier.

#define TT 2048
#define BB 32
#define II 512
#define HH 512
#define G4 2048               // 4*H
#define NT (G4 / 16)          // 128 n-tiles of gates
#define MT_TOT (TT * BB / 16) // 4096 row tiles of (t,b)
#define NWG_DIR 8
#define LDS_W_BYTES (256 * 1024)          // 256 Whh fragment tiles x 1KB
#define LDS_H_OFF   LDS_W_BYTES           // h staging area (32KB)
#define LDS_TOTAL   (LDS_W_BYTES + BB * HH * 2)

typedef __bf16 bf16_t;
typedef __attribute__((ext_vector_type(16))) __bf16 v16bf;
typedef __attribute__((ext_vector_type(8)))  float  v8f;

union FragB { v16bf v; float4 f4[2]; };
union FragC { v8f   v; float4 f4[2]; float f[8]; };

__device__ __forceinline__ float sigf(float x) { return 1.0f / (1.0f + expf(-x)); }

// async global->LDS 16B copy (CDNA5 ASYNCcnt path; no VGPR staging)
__device__ __forceinline__ void async_ld16(unsigned lds_off, const void* gptr) {
  asm volatile("global_load_async_to_lds_b128 %0, %1, off"
               :: "v"(lds_off), "v"(gptr)
               : "memory");
}
__device__ __forceinline__ void wait_async0() {
  asm volatile("s_wait_asynccnt 0" ::: "memory");
}

// ---- Phase 1: pack x into A-fragment layout (ISA 7.12.2, 16-bit A 16x32) ----
__global__ void __launch_bounds__(256) pack_x_kernel(const float* __restrict__ x,
                                                     bf16_t* __restrict__ xp) {
  size_t idx = (size_t)blockIdx.x * blockDim.x + threadIdx.x;
  int e    = idx & 15;
  int lane = (idx >> 4) & 31;
  int kt   = (idx >> 9) & 15;
  int mtile = (int)(idx >> 13);
  int row = mtile * 16 + (lane & 15);
  int k   = kt * 32 + ((lane >> 4) * 8) + ((e >> 3) * 16) + (e & 7);
  xp[idx] = (bf16_t)x[(size_t)row * II + k];
}

// ---- pack W into B-fragment layout (32x16 bf16 B tile per (nt,kt)) ---------
__global__ void __launch_bounds__(256) pack_w_kernel(const float* __restrict__ w,
                                                     bf16_t* __restrict__ wp, int kdim) {
  size_t idx = (size_t)blockIdx.x * blockDim.x + threadIdx.x;
  int e    = idx & 15;
  int lane = (idx >> 4) & 31;
  int kt   = (idx >> 9) & 15;
  int nt   = (int)(idx >> 13);
  int n = nt * 16 + (lane & 15);
  int k = kt * 32 + ((lane >> 4) * 8) + ((e >> 3) * 16) + (e & 7);
  wp[idx] = (bf16_t)w[(size_t)n * kdim + k];
}

// ---- init h state (bf16, buffer 0) and zero per-direction barriers ---------
__global__ void __launch_bounds__(256) init_state_kernel(const float* __restrict__ h0f,
                                                         const float* __restrict__ h0b,
                                                         bf16_t* __restrict__ hstate,
                                                         unsigned* __restrict__ bars) {
  int idx = blockIdx.x * blockDim.x + threadIdx.x;   // 0 .. 2*32*512-1
  int dir = idx >> 14;
  int off = idx & 16383;
  const float* src = dir ? h0b : h0f;
  hstate[(size_t)dir * 2 * (BB * HH) + off] = (bf16_t)src[off];
  if (idx < 2) bars[idx] = 0u;
}

// ---- Phase 2: x_proj = x @ Wih^T + b, stored fp32 in D-fragment order ------
__global__ void __launch_bounds__(256) xproj_kernel(const bf16_t* __restrict__ xp,
                                                    const bf16_t* __restrict__ wihp,
                                                    const float* __restrict__ bias_f,
                                                    const float* __restrict__ bias_b,
                                                    float* __restrict__ xproj_sw) {
  int lane   = threadIdx.x & 31;
  int wave   = blockIdx.x * (blockDim.x >> 5) + (threadIdx.x >> 5);
  int nwaves = gridDim.x * (blockDim.x >> 5);
  const size_t wstride = (size_t)NT * 16 * 32 * 16;
  size_t ntiles = (size_t)2 * MT_TOT * NT;
  for (size_t tile = wave; tile < ntiles; tile += nwaves) {
    int gn    = (int)(tile % NT);
    size_t r  = tile / NT;
    int mtile = (int)(r % MT_TOT);
    int dir   = (int)(r / MT_TOT);
    const float* bias = dir ? bias_b : bias_f;
    float bv = bias[gn * 16 + (lane & 15)];
    FragC acc;
#pragma unroll
    for (int i = 0; i < 8; ++i) acc.f[i] = bv;

    const bf16_t* abase = xp + (size_t)mtile * (16 * 32 * 16);
    const bf16_t* bbase = wihp + (size_t)dir * wstride + (size_t)gn * (16 * 32 * 16);
#pragma unroll 4
    for (int kt = 0; kt < 16; ++kt) {
      FragB a, b;
      const float4* ap = (const float4*)(abase + ((size_t)kt * 32 + lane) * 16);
      a.f4[0] = ap[0]; a.f4[1] = ap[1];
      const float4* bp = (const float4*)(bbase + ((size_t)kt * 32 + lane) * 16);
      b.f4[0] = bp[0]; b.f4[1] = bp[1];
      acc.v = __builtin_amdgcn_wmma_f32_16x16x32_bf16(false, a.v, false, b.v,
                                                      (short)0, acc.v, false, false);
    }
    int t  = mtile >> 1;
    int mt = mtile & 1;
    int f  = mt * NT + gn;
    float4* dp = (float4*)(xproj_sw + ((((size_t)dir * TT + t) * 256 + f) * 32 + lane) * 8);
    dp[0] = acc.f4[0]; dp[1] = acc.f4[1];
  }
}

// ---- Phase 3: persistent recurrence with LDS-resident Whh ------------------
__global__ void __launch_bounds__(256) lstm_rec_kernel(const float* __restrict__ xproj_sw,
                                                       const bf16_t* __restrict__ whhp,
                                                       bf16_t* __restrict__ hstate,
                                                       const float* __restrict__ c0f,
                                                       const float* __restrict__ c0b,
                                                       float* __restrict__ out,
                                                       unsigned* __restrict__ bars) {
  extern __shared__ char smem[];   // [0,256KB): Whh tiles; [256KB,288KB): h stage

  int dir  = blockIdx.x / NWG_DIR;
  int wg   = blockIdx.x % NWG_DIR;
  int wave = threadIdx.x >> 5;
  int lane = threadIdx.x & 31;
  int htl  = wave & 3;             // local h-tile 0..3
  int ht   = wg * 4 + htl;         // global h-tile 0..31
  int mt   = wave >> 2;            // batch tile 0..1

  const float* c0 = dir ? c0b : c0f;
  bf16_t* hbuf = hstate + (size_t)dir * 2 * (BB * HH);
  const bf16_t* wbase = whhp + (size_t)dir * ((size_t)NT * 16 * 32 * 16);
  unsigned* bar = bars + dir;

  // ---- one-time: async-load this WG's 256KB Whh slice into LDS ----
  // LDS tile ti = ((htl*4+g)*16 + kt), 1KB each; lds byte addr = j*16.
  for (int j = threadIdx.x; j < LDS_W_BYTES / 16; j += 256) {
    int q    = j & 63;             // 64 x 16B per tile
    int ti   = j >> 6;
    int l    = q >> 1;             // fragment lane 0..31
    int half = q & 1;
    int kt  = ti & 15;
    int g   = (ti >> 4) & 3;
    int hl  = ti >> 6;
    int nt  = g * 32 + wg * 4 + hl;
    const bf16_t* src = wbase + (((size_t)nt * 16 + kt) * 32 + l) * 16 + half * 8;
    async_ld16((unsigned)(j * 16), src);
  }
  wait_async0();
  __syncthreads();

  int mrow = mt * 16 + (lane & 15);        // A-operand row (batch)
  int moff = mt * 16 + 8 * (lane >> 4);    // C/D row base (add r)
  int col  = ht * 16 + (lane & 15);        // h column

  float c[8];
#pragma unroll
  for (int r = 0; r < 8; ++r) c[r] = c0[(size_t)(moff + r) * HH + col];

  for (int s = 0; s < TT; ++s) {
    int t = dir ? (TT - 1 - s) : s;
    const bf16_t* hcur = hbuf + (size_t)(s & 1) * (BB * HH);
    bf16_t* hnext      = hbuf + (size_t)((s + 1) & 1) * (BB * HH);

    // ---- async-stage h (32KB) into LDS ----
#pragma unroll
    for (int i = 0; i < 8; ++i) {
      int j = threadIdx.x + i * 256;       // 2048 x 16B
      async_ld16((unsigned)(LDS_H_OFF + j * 16), hcur + (size_t)j * 8);
    }

    // C init = x_proj fragment (fp32, D-layout) for each of the 4 gates
    const float* xb = xproj_sw + ((size_t)dir * TT + t) * ((size_t)256 * 32 * 8);
    FragC acc[4];
#pragma unroll
    for (int g = 0; g < 4; ++g) {
      int f = mt * NT + g * 32 + ht;
      const float4* cp = (const float4*)(xb + ((size_t)f * 32 + lane) * 8);
      acc[g].f4[0] = cp[0]; acc[g].f4[1] = cp[1];
    }
    // prefetch next step's x_proj fragment strip
    if (s + 1 < TT) {
      int tn = dir ? (TT - 2 - s) : (s + 1);
      __builtin_prefetch(xproj_sw + ((size_t)dir * TT + tn) * ((size_t)256 * 32 * 8)
                                  + ((size_t)(mt * NT + ht) * 32 + lane) * 8, 0, 0);
    }

    wait_async0();
    __syncthreads();   // h stage visible to all waves

    // gates += h @ Whh^T   (K = 512 -> 16 bf16 WMMAs per gate tile), all LDS
    const char* hs = smem + LDS_H_OFF + (size_t)mrow * (HH * 2);
#pragma unroll 2
    for (int kt = 0; kt < 16; ++kt) {
      FragB a;
      const float4* ap = (const float4*)(hs + (kt * 32 + (lane >> 4) * 8) * 2);
      a.f4[0] = ap[0];   // k-group 0 (8 bf16)
      a.f4[1] = ap[2];   // k-group +16 (8 bf16)
#pragma unroll
      for (int g = 0; g < 4; ++g) {
        FragB b;
        const float4* bp =
            (const float4*)(smem + ((size_t)((htl * 4 + g) * 16 + kt)) * 1024 + lane * 32);
        b.f4[0] = bp[0]; b.f4[1] = bp[1];
        acc[g].v = __builtin_amdgcn_wmma_f32_16x16x32_bf16(false, a.v, false, b.v,
                                                           (short)0, acc[g].v, false, false);
      }
    }

    // elementwise LSTM cell; c stays in registers
#pragma unroll
    for (int r = 0; r < 8; ++r) {
      float iv = sigf(acc[0].f[r]);
      float fv = sigf(acc[1].f[r]);
      float gv = tanhf(acc[2].f[r]);
      float ov = sigf(acc[3].f[r]);
      float cn = fv * c[r] + iv * gv;
      c[r] = cn;
      float hv = ov * tanhf(cn);
      int m = moff + r;
      out[((size_t)t * BB + m) * (2 * HH) + (size_t)dir * HH + col] = hv;
      hnext[(size_t)m * HH + col] = (bf16_t)hv;
    }

    // per-direction grid barrier (8 WGs, all resident); also guards the LDS
    // h-stage against being overwritten while any wave still reads it.
    __syncthreads();
    if (threadIdx.x == 0) {
      __threadfence();
      __hip_atomic_fetch_add(bar, 1u, __ATOMIC_RELEASE, __HIP_MEMORY_SCOPE_AGENT);
      unsigned target = (unsigned)(NWG_DIR * (s + 1));
      while (__hip_atomic_load(bar, __ATOMIC_ACQUIRE, __HIP_MEMORY_SCOPE_AGENT) < target) {
        __builtin_amdgcn_s_sleep(1);
      }
    }
    __syncthreads();
  }
}

extern "C" void kernel_launch(void* const* d_in, const int* in_sizes, int n_in,
                              void* d_out, int out_size, void* d_ws, size_t ws_size,
                              hipStream_t stream) {
  const float* x     = (const float*)d_in[0];
  const float* h0_f  = (const float*)d_in[1];
  const float* c0_f  = (const float*)d_in[2];
  const float* h0_b  = (const float*)d_in[3];
  const float* c0_b  = (const float*)d_in[4];
  const float* Wih_f = (const float*)d_in[5];
  const float* Whh_f = (const float*)d_in[6];
  const float* b_f   = (const float*)d_in[7];
  const float* Wih_b = (const float*)d_in[8];
  const float* Whh_b = (const float*)d_in[9];
  const float* b_b   = (const float*)d_in[10];
  float* out = (float*)d_out;

  size_t off = 0;
  auto take = [&](size_t bytes) {
    size_t o = off;
    off += (bytes + 255) & ~(size_t)255;
    return o;
  };
  const size_t XP_ELEMS = (size_t)MT_TOT * 16 * 32 * 16;      // 33,554,432 bf16
  const size_t WP_ELEMS = (size_t)NT * 16 * 32 * 16;          // 1,048,576 bf16 / matrix
  const size_t XPROJ_FLOATS = (size_t)2 * TT * 256 * 32 * 8;  // 268,435,456 f32

  char* ws = (char*)d_ws;
  bf16_t*   xp      = (bf16_t*)(ws + take(XP_ELEMS * sizeof(bf16_t)));
  bf16_t*   wihp    = (bf16_t*)(ws + take(2 * WP_ELEMS * sizeof(bf16_t)));
  bf16_t*   whhp    = (bf16_t*)(ws + take(2 * WP_ELEMS * sizeof(bf16_t)));
  float*    xprojsw = (float*)(ws + take(XPROJ_FLOATS * sizeof(float)));
  bf16_t*   hstate  = (bf16_t*)(ws + take((size_t)2 * 2 * BB * HH * sizeof(bf16_t)));
  unsigned* bars    = (unsigned*)(ws + take(256));

  pack_x_kernel<<<(int)(XP_ELEMS / 256), 256, 0, stream>>>(x, xp);
  pack_w_kernel<<<(int)(WP_ELEMS / 256), 256, 0, stream>>>(Wih_f, wihp, II);
  pack_w_kernel<<<(int)(WP_ELEMS / 256), 256, 0, stream>>>(Wih_b, wihp + WP_ELEMS, II);
  pack_w_kernel<<<(int)(WP_ELEMS / 256), 256, 0, stream>>>(Whh_f, whhp, HH);
  pack_w_kernel<<<(int)(WP_ELEMS / 256), 256, 0, stream>>>(Whh_b, whhp + WP_ELEMS, HH);
  init_state_kernel<<<(2 * BB * HH) / 256, 256, 0, stream>>>(h0_f, h0_b, hstate, bars);
  xproj_kernel<<<1024, 256, 0, stream>>>(xp, wihp, b_f, b_b, xprojsw);
  lstm_rec_kernel<<<2 * NWG_DIR, 256, LDS_TOTAL, stream>>>(xprojsw, whhp, hstate,
                                                           c0_f, c0_b, out, bars);
}
